// SyntheticGraphLearner_36825049596050
// MI455X (gfx1250) — matrix operand order repaired and checked by reading, not compiled, not run on hardware
//
#include <hip/hip_runtime.h>
#include <stdint.h>

typedef __attribute__((ext_vector_type(16))) __bf16 v16bf;
typedef __attribute__((ext_vector_type(8)))  float  v8f;
typedef __attribute__((ext_vector_type(4)))  unsigned int u32x4;
typedef __attribute__((ext_vector_type(4)))  int    i32x4;
typedef __attribute__((ext_vector_type(8)))  int    i32x8;

#define B_     4
#define N_     64
#define FIN_   128
#define H_     8
#define HID_   128
#define KSPLIT 8
#define KCHUNK (N_ / KSPLIT)
#define NEG_SLOPE 0.2f

// ws layout (bf16 elements)
#define WS_X    0                       // [4][64][128] row-major
#define WS_WPRE (WS_X + B_*N_*FIN_)     // fragment-linear, 4 slabs x 4096
#define WS_W1   (WS_WPRE + FIN_*HID_)   // per h: 8 slabs x 4096 (Wa slabs 0-3, Wb 4-7)

// ---------------- Tensor Data Mover: linear (optionally padded) LDS fill ----
// Descriptor per CDNA5 ISA ch.8: group0 = {count, lds_addr, global_addr, type=2},
// group1 = {data_size=4B, pad ctl, tensor_dim0=tile_dim0=ndw, dim1=1, stride=ndw}.
__device__ __forceinline__ void tdm_load_linear(void* lds_ptr, const void* gptr,
                                                uint32_t ndw, uint32_t pad_en,
                                                uint32_t pad_int, uint32_t pad_amt) {
  const uint64_t ga = (uint64_t)(uintptr_t)gptr;
  const uint32_t la = (uint32_t)(uintptr_t)lds_ptr;  // low 32 bits = LDS offset
  u32x4 g0;
  g0.x = 1u;                                           // count=1, gather off
  g0.y = la;                                           // lds_addr
  g0.z = (uint32_t)ga;                                 // global_addr[31:0]
  g0.w = (uint32_t)((ga >> 32) & 0x1FFFFFFu) | (2u << 30);  // addr[56:32], type=2
  i32x8 g1;
  g1[0] = (int)((2u << 16) | (pad_en << 20) | (pad_int << 22) | (pad_amt << 25));
  g1[1] = (int)((ndw & 0xFFFFu) << 16);                // tensor_dim0[15:0] @bit48
  g1[2] = (int)((ndw >> 16) | (1u << 16));             // tensor_dim0 hi, tensor_dim1=1
  g1[3] = (int)((ndw & 0xFFFFu) << 16);                // tile_dim0 @bit112
  g1[4] = 0;                                           // tile_dim1/2 unused
  g1[5] = (int)ndw;                                    // tensor_dim0_stride lo32
  g1[6] = 0;
  g1[7] = 0;
  const i32x4 z4 = {0, 0, 0, 0};
#if __clang_major__ >= 23
  const i32x8 z8 = {0, 0, 0, 0, 0, 0, 0, 0};
  __builtin_amdgcn_tensor_load_to_lds(g0, g1, z4, z4, z8, 0);
#else
  __builtin_amdgcn_tensor_load_to_lds(g0, g1, z4, z4, 0);
#endif
}

// ---------------- WMMA fragment loaders (ISA 7.12.2, 16-bit, wave32) --------
// A 16x32, this code's own contiguous k order (B matches): lanes 0-15 -> M=lane, K {0..15}; lanes 16-31 -> K {16..31}.
__device__ __forceinline__ v16bf load_a_frag(const __bf16* A, int lda,
                                             int mr, int kb, int lane) {
  v16bf a;
  const int hf = lane >> 4;
  const __bf16* row = A + (size_t)(mr + (lane & 15)) * lda;
#pragma unroll
  for (int e = 0; e < 8; ++e) a[e] = row[kb + hf * 16 + e];
#pragma unroll
  for (int e = 0; e < 8; ++e) a[8 + e] = row[kb + hf * 16 + 8 + e];
  return a;
}

// B 32x16 from fragment-linear slab [n][kk]: lane reads 16 contiguous bf16.
__device__ __forceinline__ v16bf load_b_frag_lin(const __bf16* slab, int nc, int lane) {
  const int n  = nc + (lane & 15);
  const int hf = lane >> 4;
  return *(const v16bf*)(slab + (size_t)n * 32 + hf * 16);  // 32B aligned
}

// C(64x128) = A(LDS bf16, pitch 136) @ W(fragment-linear bf16 global, 4 slabs),
// slabs pulled by TDM into LDS; wave w owns column strip w*16.
__device__ __forceinline__ void gemm64x128(const __bf16* Alds,
                                           const __bf16* __restrict__ WgFrag,
                                           __bf16* slab,
                                           int nc, int lane, int tid, v8f acc[4]) {
  for (int s = 0; s < 4; ++s) {
    if (s < 3)  // prefetch next slab into GL2 (global_prefetch_b8)
      __builtin_prefetch(WgFrag + (size_t)(s + 1) * 4096 + tid * 16, 0, 3);
    if (tid < 32) {  // wave 0 drives the Tensor Data Mover
      tdm_load_linear(slab, WgFrag + (size_t)s * 4096, 2048, 0, 0, 0);
      __builtin_amdgcn_s_wait_tensorcnt(0);
    }
    __syncthreads();
    const v16bf bfrag = load_b_frag_lin(slab, nc, lane);
#pragma unroll
    for (int mt = 0; mt < 4; ++mt) {
      const v16bf afrag = load_a_frag(Alds, 136, mt * 16, s * 32, lane);
      acc[mt] = __builtin_amdgcn_wmma_f32_16x16x32_bf16(
          false, afrag, false, bfrag, (short)0, acc[mt], false, false);
    }
    __syncthreads();
  }
}

// ---------------- prep: fp32 -> bf16 + B-fragment swizzle into workspace ----
// Fragment-linear: elem(slab s, n, kk) = W[(s*32+kk)*128 + n], flat = s*4096+n*32+kk.
__global__ __launch_bounds__(256)
void prep_convert(const float* __restrict__ X, const float* __restrict__ Wpre,
                  const float* __restrict__ W1, __bf16* __restrict__ ws) {
  const int stride = gridDim.x * blockDim.x;
  const int idx = blockIdx.x * blockDim.x + threadIdx.x;
  for (int e = idx; e < B_ * N_ * FIN_; e += stride)
    ws[WS_X + e] = (__bf16)X[e];
  for (int e = idx; e < FIN_ * HID_; e += stride) {
    const int s = e >> 12, r = e & 4095, n = r >> 5, kk = r & 31;
    ws[WS_WPRE + e] = (__bf16)Wpre[(s * 32 + kk) * HID_ + n];
  }
  for (int e = idx; e < H_ * 2 * FIN_ * HID_; e += stride) {
    const int h = e >> 15, r = e & 32767;
    const int s = r >> 12, rr = r & 4095, n = rr >> 5, kk = rr & 31;
    ws[WS_W1 + e] = (__bf16)W1[(size_t)h * 2 * FIN_ * HID_ + (s * 32 + kk) * HID_ + n];
  }
}

// ---------------- fused main kernel ----------------------------------------
__global__ __launch_bounds__(256)
void graph_learner_fused(const __bf16* __restrict__ ws,
                         const float* __restrict__ bpre,
                         const float* __restrict__ b1,
                         const float* __restrict__ W2,
                         const float* __restrict__ b2,
                         float* __restrict__ out) {
  __shared__ __align__(32) __bf16 Xs[N_][136];   // X[b] then v; pitch via TDM pad
  __shared__ __align__(32) __bf16 Wslab[32 * 128];  // fragment-linear slab
  __shared__ __bf16 AiT[HID_][72];               // Ai^T (+b1 folded): [m][i]
  __shared__ __bf16 BkS[N_][136];                // Bk: [k][m]
  __shared__ float  W2s[HID_];

  const int tid  = threadIdx.x;
  const int lane = tid & 31;
  const int wv   = tid >> 5;
  const int nc   = wv * 16;
  const int nl   = nc + (lane & 15);
  const int hf   = lane >> 4;

  const int blk = blockIdx.x;
  const int b   = blk / (H_ * KSPLIT);
  const int hr  = blk % (H_ * KSPLIT);
  const int h   = hr / KSPLIT;
  const int sp  = hr % KSPLIT;

  // TDM: X[b] (bf16, 4096 dwords) -> Xs with 4-dword pad per 64-dword row
  // (pad_interval code 5 = 64 dw, pad_amount code 3 = 4 dw) => pitch 136 bf16.
  if (tid < 32) {
    tdm_load_linear(&Xs[0][0], ws + WS_X + (size_t)b * N_ * FIN_, 4096, 1, 5, 3);
    __builtin_amdgcn_s_wait_tensorcnt(0);
  }
  if (tid < HID_) W2s[tid] = W2[h * HID_ + tid];
  // (first barrier inside gemm64x128 publishes Xs/W2s)

  // ---- GEMM1: v = X @ W_pre + b_pre ----
  v8f accV[4] = {};
  gemm64x128(&Xs[0][0], ws + WS_WPRE, Wslab, nc, lane, tid, accV);
  {
    const float bp = bpre[nl];
#pragma unroll
    for (int mt = 0; mt < 4; ++mt)
#pragma unroll
      for (int r = 0; r < 8; ++r)
        Xs[mt * 16 + hf * 8 + r][nl] = (__bf16)(accV[mt][r] + bp);
  }
  __syncthreads();

  // ---- GEMM2: Ai = v @ Wa[h] + b1[h] (stored transposed) ----
  const __bf16* WaF = ws + WS_W1 + (size_t)h * 2 * FIN_ * HID_;
  v8f accA[4] = {};
  gemm64x128(&Xs[0][0], WaF, Wslab, nc, lane, tid, accA);
  {
    const float bb = b1[h * HID_ + nl];
#pragma unroll
    for (int mt = 0; mt < 4; ++mt)
#pragma unroll
      for (int r = 0; r < 8; ++r)
        AiT[nl][mt * 16 + hf * 8 + r] = (__bf16)(accA[mt][r] + bb);
  }

  // ---- GEMM3: Bk = v @ Wb[h] ----
  v8f accB[4] = {};
  gemm64x128(&Xs[0][0], WaF + FIN_ * HID_, Wslab, nc, lane, tid, accB);
#pragma unroll
  for (int mt = 0; mt < 4; ++mt)
#pragma unroll
    for (int r = 0; r < 8; ++r)
      BkS[mt * 16 + hf * 8 + r][nl] = (__bf16)accB[mt][r];
  __syncthreads();

  // ---- Pairwise leaky-MLP reduction + fused permute/pad store -------------
  // S[i,k] = sum_m leaky(AiT[m][i] + BkS[k][m]) * W2[m] + b2[h]
  // out[b,h, k-(k>i), i] = S[i,k] (k != i); row N-1 zero-padded.
  const float b2v = b2[h];
  float* outp = out + (size_t)(b * H_ + h) * N_ * N_;
  for (int p = tid; p < KCHUNK * N_; p += 256) {
    const int k = sp * KCHUNK + (p >> 6);  // uniform per wave -> LDS broadcast
    const int i = p & 63;                  // contiguous per lane
    float s = b2v;
#pragma unroll 8
    for (int m = 0; m < HID_; ++m) {
      float x = (float)AiT[m][i] + (float)BkS[k][m];
      x = (x >= 0.0f) ? x : NEG_SLOPE * x;
      s += x * W2s[m];
    }
    if (k != i) {
      const int j = k - (k > i ? 1 : 0);
      outp[j * N_ + i] = s;
    }
  }
  if (sp == 0)
    for (int i2 = tid; i2 < N_; i2 += 256)
      outp[(N_ - 1) * N_ + i2] = 0.0f;
}

extern "C" void kernel_launch(void* const* d_in, const int* in_sizes, int n_in,
                              void* d_out, int out_size, void* d_ws, size_t ws_size,
                              hipStream_t stream) {
  const float* X    = (const float*)d_in[0];
  const float* Wpre = (const float*)d_in[1];
  const float* bpre = (const float*)d_in[2];
  const float* W1   = (const float*)d_in[3];
  const float* b1   = (const float*)d_in[4];
  const float* W2   = (const float*)d_in[5];
  const float* b2   = (const float*)d_in[6];
  float* out = (float*)d_out;
  __bf16* ws = (__bf16*)d_ws;  // needs ~610 KB: bf16 X + swizzled W_pre/W1

  prep_convert<<<dim3(128), dim3(256), 0, stream>>>(X, Wpre, W1, ws);
  graph_learner_fused<<<dim3(B_ * H_ * KSPLIT), dim3(256), 0, stream>>>(
      ws, bpre, b1, W2, b2, out);
}
